// TransformerLayer_66760971649653
// MI455X (gfx1250) — matrix operand order repaired and checked by reading, not compiled, not run on hardware
//
#include <hip/hip_runtime.h>
#include <stdint.h>

// ---------------------------------------------------------------------------
// Types for CDNA5 WMMA (wave32): v16bf A/B fragments, v8f accumulator.
// ---------------------------------------------------------------------------
typedef unsigned short bf16_t;
typedef __attribute__((ext_vector_type(16))) __bf16 v16bf;
typedef __attribute__((ext_vector_type(8)))  float  v8f;

constexpr int DM  = 512;  // d_model
constexpr int NH  = 8;    // heads
constexpr int DHD = 64;   // head dim
constexpr int NB  = 8;    // batch
constexpr int SL  = 256;  // seq len
constexpr int KW  = 9;    // conv kernel width
constexpr int PD  = 4;    // conv padding
constexpr float EPSV = 1e-5f;

__device__ __forceinline__ bf16_t f2bf(float f) {
  uint32_t u = __float_as_uint(f);
  uint32_t r = u + 0x7FFFu + ((u >> 16) & 1u);  // round-to-nearest-even
  return (bf16_t)(r >> 16);
}

union FragBF { v16bf v; uint4 q[2]; };
union FragF  { v8f v; float f[8]; };

// A fragment (16x32 bf16, row-major src, lane-pre-offset pointer):
//   lanes 0-15 : row = m0+lane,    K 0..15
//   lanes 16-31: row = m0+lane-16, K 16..31
__device__ __forceinline__ v16bf load_a(const bf16_t* p) {
  FragBF f;
  f.q[0] = *(const uint4*)(p);
  f.q[1] = *(const uint4*)(p + 8);
  return f.v;
}
// B fragment from N x K (pre-transposed) memory:
//   lane holds column n = n0+(lane&15), 16 consecutive K (lanes>=16: +16)
__device__ __forceinline__ v16bf load_b(const bf16_t* p) {
  FragBF f;
  f.q[0] = *(const uint4*)(p);
  f.q[1] = *(const uint4*)(p + 8);
  return f.v;
}
__device__ __forceinline__ v8f wmma_bf16(v16bf a, v16bf b, v8f c) {
  return __builtin_amdgcn_wmma_f32_16x16x32_bf16(false, a, false, b,
                                                 (short)0, c, false, false);
}

// ---------------------------------------------------------------------------
// Generic batched bf16 GEMM:  C[b,h] (+)= A[b,h] (MxK) * BT[b,h]^T (N x K)
// Block = 256 threads = 8 waves; block tile 128x128; wave tile 32x64
// (2 M-frags x 4 N-frags of 16x16, fp32 accumulate).
//
// REQUIREMENTS (hold for every call site here): M % 32 == 0 within the grid,
// N % 64 == 0, K % 32 == 0.  This lets each wave either fully own an aligned
// 32x64 strip or exit, so there is NO lane-divergent predication anywhere in
// the WMMA loop (EXEC stays all-1s, per ISA requirement for WMMA).
//
// flags: bit0 = accumulate into existing Cf, bit1 = relu epilogue.
// ---------------------------------------------------------------------------
__global__ __launch_bounds__(256) void gemm_bf16_kernel(
    const bf16_t* __restrict__ A, long long sAb, long long sAh, int lda,
    const bf16_t* __restrict__ BT, long long sBb, long long sBh, int ldb,
    const float* __restrict__ bias,
    float* __restrict__ Cf, bf16_t* __restrict__ Cb,
    long long sCb, long long sCh, int ldc,
    int M, int N, int K, int Hdim, int flags)
{
  const int z = blockIdx.z;
  const int bb = z / Hdim, hh = z - bb * Hdim;
  const bf16_t* Abase = A  + (size_t)bb * sAb + (size_t)hh * sAh;
  const bf16_t* Bbase = BT + (size_t)bb * sBb + (size_t)hh * sBh;

  const int w = threadIdx.x >> 5;
  const int lane = threadIdx.x & 31;
  const int wm = w & 3, wn = w >> 2;
  const int m0 = blockIdx.x * 128 + wm * 32;
  const int n0 = blockIdx.y * 128 + wn * 64;
  if (m0 >= M || n0 >= N) return;  // wave-uniform; tiles inside are full

  FragF acc[2][4];
#pragma unroll
  for (int i = 0; i < 2; ++i)
#pragma unroll
    for (int j = 0; j < 4; ++j)
#pragma unroll
      for (int r = 0; r < 8; ++r) acc[i][j].f[r] = 0.f;

  const int rA  = lane & 15;
  const int kaA = (lane & 16) ? 16 : 0;  // A lane K-offset
  const int kbB = (lane & 16) ? 16 : 0;  // B lane K-offset
  const bf16_t* a0p = Abase + (size_t)(m0 + rA) * lda + kaA;
  const bf16_t* a1p = a0p + (size_t)16 * lda;
  const bf16_t* bp0 = Bbase + (size_t)(n0 + rA) * ldb + kbB;
  const bf16_t* bp1 = bp0 + (size_t)16 * ldb;
  const bf16_t* bp2 = bp0 + (size_t)32 * ldb;
  const bf16_t* bp3 = bp0 + (size_t)48 * ldb;

  for (int k0 = 0; k0 < K; k0 += 32) {
    // prefetch next K-tile (global_prefetch_b8); harmless on last iter
    __builtin_prefetch(a0p + k0 + 32, 0, 3);
    __builtin_prefetch(bp0 + k0 + 32, 0, 3);

    v16bf a0 = load_a(a0p + k0);
    v16bf a1 = load_a(a1p + k0);
    v16bf b0 = load_b(bp0 + k0);
    v16bf b1 = load_b(bp1 + k0);
    v16bf b2 = load_b(bp2 + k0);
    v16bf b3 = load_b(bp3 + k0);
    acc[0][0].v = wmma_bf16(a0, b0, acc[0][0].v);
    acc[1][0].v = wmma_bf16(a1, b0, acc[1][0].v);
    acc[0][1].v = wmma_bf16(a0, b1, acc[0][1].v);
    acc[1][1].v = wmma_bf16(a1, b1, acc[1][1].v);
    acc[0][2].v = wmma_bf16(a0, b2, acc[0][2].v);
    acc[1][2].v = wmma_bf16(a1, b2, acc[1][2].v);
    acc[0][3].v = wmma_bf16(a0, b3, acc[0][3].v);
    acc[1][3].v = wmma_bf16(a1, b3, acc[1][3].v);
  }

  // C fragment layout: lane holds column n=(lane&15); vgpr r -> row r (+8 for hi lanes)
  const int mo = (lane & 16) ? 8 : 0;
  const int nc = lane & 15;
#pragma unroll
  for (int i = 0; i < 2; ++i)
#pragma unroll
    for (int j = 0; j < 4; ++j) {
      const int nn = n0 + j * 16 + nc;
      const float bv = bias ? bias[nn] : 0.f;
#pragma unroll
      for (int r = 0; r < 8; ++r) {
        const int mm = m0 + i * 16 + mo + r;
        size_t idx = (size_t)bb * sCb + (size_t)hh * sCh + (size_t)mm * ldc + nn;
        float v = acc[i][j].f[r] + bv;
        if (flags & 1) v += Cf[idx];
        if (flags & 2) v = fmaxf(v, 0.f);
        if (Cf) Cf[idx] = v;
        if (Cb) Cb[idx] = f2bf(v);
      }
    }
}

// ---------------------------------------------------------------------------
// BD score kernel: bd[b,h,q,n] = sum_d (q+v)[b,q,h,d] * r[q,n,h,d]
// One wave per (q, h, 64-wide n tile): M=16 (rows = batch, rows>=8 discarded),
// N=64, K=64.  Adds into the AC scores buffer.  All WMMAs execute with full
// EXEC; the only divergent code (lane<16 stores) comes after them.
// ---------------------------------------------------------------------------
__global__ __launch_bounds__(32) void bd_kernel(
    const bf16_t* __restrict__ qv,   // [NB, SL, DM] bf16
    const bf16_t* __restrict__ rmat, // [SL, SL, DM] bf16
    float* __restrict__ scores)      // [NB, NH, SL, SL] f32
{
  const int q = blockIdx.x, h = blockIdx.y, nt = blockIdx.z;
  const int lane = threadIdx.x;
  const int row = lane & 15;
  const int bcl = row < NB ? row : (NB - 1);  // clamp pad rows
  const int kaA = (lane & 16) ? 16 : 0;
  const int kbB = (lane & 16) ? 16 : 0;
  const bf16_t* ap = qv + ((size_t)bcl * SL + q) * DM + h * DHD + kaA;
  const bf16_t* bp = rmat + ((size_t)q * SL + nt * 64 + (lane & 15)) * DM + h * DHD + kbB;

  FragF acc[4];
#pragma unroll
  for (int j = 0; j < 4; ++j)
#pragma unroll
    for (int r = 0; r < 8; ++r) acc[j].f[r] = 0.f;

#pragma unroll
  for (int k0 = 0; k0 < DHD; k0 += 32) {
    v16bf a = load_a(ap + k0);
#pragma unroll
    for (int j = 0; j < 4; ++j) {
      v16bf bf = load_b(bp + (size_t)j * 16 * DM + k0);
      acc[j].v = wmma_bf16(a, bf, acc[j].v);
    }
  }

  if (lane < 16) {  // rows 0..7 == valid batch indices
#pragma unroll
    for (int j = 0; j < 4; ++j) {
      const int n = nt * 64 + j * 16 + lane;
#pragma unroll
      for (int r = 0; r < NB; ++r) {
        size_t idx = (((size_t)r * NH + h) * SL + q) * SL + n;
        scores[idx] += acc[j].f[r];
      }
    }
  }
}

// ---------------------------------------------------------------------------
// Softmax over the key axis (256), one wave per row, scale = 1/sqrt(64).
// ---------------------------------------------------------------------------
__global__ __launch_bounds__(256) void softmax_kernel(
    const float* __restrict__ sc, bf16_t* __restrict__ at)
{
  const int row = blockIdx.x * 8 + (threadIdx.x >> 5);
  const int lane = threadIdx.x & 31;
  const float* p = sc + (size_t)row * SL;
  float x[8];
  float mx = -3.4e38f;
#pragma unroll
  for (int j = 0; j < 8; ++j) { x[j] = p[lane + j * 32] * 0.125f; mx = fmaxf(mx, x[j]); }
#pragma unroll
  for (int m = 16; m >= 1; m >>= 1) mx = fmaxf(mx, __shfl_xor(mx, m, 32));
  float s = 0.f;
#pragma unroll
  for (int j = 0; j < 8; ++j) { x[j] = __expf(x[j] - mx); s += x[j]; }
#pragma unroll
  for (int m = 16; m >= 1; m >>= 1) s += __shfl_xor(s, m, 32);
  const float inv = 1.f / s;
  bf16_t* o = at + (size_t)row * SL;
#pragma unroll
  for (int j = 0; j < 8; ++j) o[lane + j * 32] = f2bf(x[j] * inv);
}

// ---------------------------------------------------------------------------
// y = LayerNorm(base + v) * g + be ; writes fp32 and optional bf16 copies.
// One wave per 512-wide row.
// ---------------------------------------------------------------------------
__global__ __launch_bounds__(256) void ln_kernel(
    const float* __restrict__ base, const float* __restrict__ y,
    const float* __restrict__ g, const float* __restrict__ be,
    float* __restrict__ outf, bf16_t* __restrict__ outb)
{
  const int row = blockIdx.x * 8 + (threadIdx.x >> 5);
  const int lane = threadIdx.x & 31;
  const float* pb = base + (size_t)row * DM;
  const float* py = y + (size_t)row * DM;
  float v[16];
  float s = 0.f;
#pragma unroll
  for (int j = 0; j < 16; ++j) { v[j] = pb[lane + j * 32] + py[lane + j * 32]; s += v[j]; }
#pragma unroll
  for (int m = 16; m >= 1; m >>= 1) s += __shfl_xor(s, m, 32);
  const float mean = s * (1.f / DM);
  float var = 0.f;
#pragma unroll
  for (int j = 0; j < 16; ++j) { float d = v[j] - mean; var += d * d; }
#pragma unroll
  for (int m = 16; m >= 1; m >>= 1) var += __shfl_xor(var, m, 32);
  const float inv = rsqrtf(var * (1.f / DM) + EPSV);
#pragma unroll
  for (int j = 0; j < 16; ++j) {
    const int c = lane + j * 32;
    const float o = (v[j] - mean) * inv * g[c] + be[c];
    outf[(size_t)row * DM + c] = o;
    if (outb) outb[(size_t)row * DM + c] = f2bf(o);
  }
}

// --------------------------- elementwise helpers ---------------------------
__global__ void cast_mod_kernel(const float* __restrict__ src,
                                bf16_t* __restrict__ dst, size_t n, size_t mod) {
  size_t i = (size_t)blockIdx.x * 256 + threadIdx.x;
  if (i >= n) return;
  dst[i] = f2bf(src[i % mod]);
}
__global__ void row_bias_cast_kernel(const float* __restrict__ src,
                                     const float* __restrict__ bias,
                                     bf16_t* __restrict__ dst, size_t n) {
  size_t i = (size_t)blockIdx.x * 256 + threadIdx.x;
  if (i >= n) return;
  dst[i] = f2bf(src[i] + bias[i & (DM - 1)]);
}
// W (KxN fp32, row-major) -> WT (NxK bf16)
__global__ void wt_kernel(const float* __restrict__ w, bf16_t* __restrict__ wt) {
  size_t i = (size_t)blockIdx.x * 256 + threadIdx.x;
  if (i >= (size_t)DM * DM) return;
  const int n = (int)(i >> 9), k = (int)(i & 511);
  wt[i] = f2bf(w[(size_t)k * DM + n]);
}
// conv weight [out,in,KW] fp32 -> per-tap [KW][out][in] bf16 (NxK form)
__global__ void convpack_kernel(const float* __restrict__ w, bf16_t* __restrict__ wp) {
  size_t i = (size_t)blockIdx.x * 256 + threadIdx.x;
  if (i >= (size_t)KW * DM * DM) return;
  const int tap = (int)(i / ((size_t)DM * DM));
  const int rem = (int)(i % ((size_t)DM * DM));
  const int co = rem >> 9, ci = rem & 511;
  wp[i] = f2bf(w[((size_t)co * DM + ci) * KW + tap]);
}
// v [b,k,h,d] bf16 -> vT [b,h,d,k] bf16
__global__ void vtr_kernel(const bf16_t* __restrict__ v, bf16_t* __restrict__ vt) {
  size_t i = (size_t)blockIdx.x * 256 + threadIdx.x;
  if (i >= (size_t)NB * NH * DHD * SL) return;
  const int k = (int)(i % SL);
  const int d = (int)((i / SL) % DHD);
  const int h = (int)((i / ((size_t)SL * DHD)) % NH);
  const int b = (int)(i / ((size_t)SL * DHD * NH));
  vt[i] = v[((size_t)(b * SL + k)) * DM + h * DHD + d];
}
// x [B,SL,DM] f32 -> zero-padded bf16 [B,SL+2*PD,DM] (optional relu)
__global__ void padcast_kernel(const float* __restrict__ x,
                               bf16_t* __restrict__ xp, int relu) {
  size_t i = (size_t)blockIdx.x * 256 + threadIdx.x;
  const size_t tot = (size_t)NB * (SL + 2 * PD) * DM;
  if (i >= tot) return;
  const int c = (int)(i % DM);
  const int s = (int)((i / DM) % (SL + 2 * PD));
  const int b = (int)(i / ((size_t)DM * (SL + 2 * PD)));
  float v = 0.f;
  if (s >= PD && s < SL + PD) {
    v = x[((size_t)b * SL + (s - PD)) * DM + c];
    if (relu) v = fmaxf(v, 0.f);
  }
  xp[i] = f2bf(v);
}

// ---------------------------------------------------------------------------
// Host-side orchestration
// ---------------------------------------------------------------------------
extern "C" void kernel_launch(void* const* d_in, const int* in_sizes, int n_in,
                              void* d_out, int out_size, void* d_ws, size_t ws_size,
                              hipStream_t stream) {
  (void)in_sizes; (void)out_size; (void)ws_size;

  const float* IN[64];
  for (int i = 0; i < n_in && i < 64; ++i) IN[i] = (const float*)d_in[i];

  char* ws = (char*)d_ws;
  size_t off = 0;
  auto alloc = [&](size_t bytes) -> void* {
    void* p = ws + off;
    off = (off + bytes + 255) & ~(size_t)255;
    return p;
  };
  const size_t TOK = (size_t)NB * SL;  // 2048 rows

  bf16_t* posw = (bf16_t*)alloc((size_t)SL * SL * DM * 2);       // pos_emb bf16
  bf16_t* rbuf = (bf16_t*)alloc((size_t)SL * SL * DM * 2);       // r projection bf16
  bf16_t* wT   = (bf16_t*)alloc((size_t)DM * DM * 2);            // weight^T scratch
  bf16_t* w1p  = (bf16_t*)alloc((size_t)KW * DM * DM * 2);
  bf16_t* w2p  = (bf16_t*)alloc((size_t)KW * DM * DM * 2);
  float*  qf   = (float*)alloc(TOK * DM * 4);
  bf16_t* qu   = (bf16_t*)alloc(TOK * DM * 2);
  bf16_t* qvb  = (bf16_t*)alloc(TOK * DM * 2);
  bf16_t* kb   = (bf16_t*)alloc(TOK * DM * 2);
  bf16_t* vb   = (bf16_t*)alloc(TOK * DM * 2);
  bf16_t* vt   = (bf16_t*)alloc(TOK * DM * 2);
  float*  sc   = (float*)alloc((size_t)NB * NH * SL * SL * 4);
  bf16_t* at   = (bf16_t*)alloc((size_t)NB * NH * SL * SL * 2);
  bf16_t* ctx  = (bf16_t*)alloc(TOK * DM * 2);
  float*  of   = (float*)alloc(TOK * DM * 4);
  bf16_t* decb = (bf16_t*)alloc(TOK * DM * 2);
  bf16_t* encb = (bf16_t*)alloc(TOK * DM * 2);
  bf16_t* dpb  = (bf16_t*)alloc(TOK * DM * 2);
  float*  x1f  = (float*)alloc(TOK * DM * 4);
  bf16_t* x1b  = (bf16_t*)alloc(TOK * DM * 2);
  float*  x2f  = (float*)alloc(TOK * DM * 4);
  bf16_t* x2b  = (bf16_t*)alloc(TOK * DM * 2);
  float*  x3f  = (float*)alloc(TOK * DM * 4);
  bf16_t* xpad = (bf16_t*)alloc((size_t)NB * (SL + 2 * PD) * DM * 2);
  float*  h1f  = (float*)alloc(TOK * DM * 4);
  bf16_t* h1p  = (bf16_t*)alloc((size_t)NB * (SL + 2 * PD) * DM * 2);
  float*  h2f  = (float*)alloc(TOK * DM * 4);

  auto ew = [](size_t n) { return dim3((unsigned)((n + 255) / 256)); };

  auto gemm = [&](const bf16_t* A, long long sAb, long long sAh, int lda,
                  const bf16_t* BT, long long sBb, long long sBh, int ldb,
                  const float* bias, float* Cf, bf16_t* Cb,
                  long long sCb, long long sCh, int ldc,
                  int M, int N, int Kd, int nb, int nh, int flags) {
    dim3 g((unsigned)((M + 127) / 128), (unsigned)((N + 127) / 128),
           (unsigned)(nb * nh));
    gemm_bf16_kernel<<<g, 256, 0, stream>>>(A, sAb, sAh, lda, BT, sBb, sBh, ldb,
                                            bias, Cf, Cb, sCb, sCh, ldc,
                                            M, N, Kd, nh, flags);
  };

  // ---- input casts ----
  cast_mod_kernel<<<ew(TOK * DM), 256, 0, stream>>>(IN[0], decb, TOK * DM, TOK * DM);
  cast_mod_kernel<<<ew(TOK * DM), 256, 0, stream>>>(IN[1], encb, TOK * DM, TOK * DM);
  cast_mod_kernel<<<ew((size_t)SL * SL * DM), 256, 0, stream>>>(
      IN[2], posw, (size_t)SL * SL * DM, (size_t)SL * SL * DM);
  // dec_pos_emb broadcast across batch
  cast_mod_kernel<<<ew(TOK * DM), 256, 0, stream>>>(IN[3], dpb, TOK * DM, (size_t)SL * DM);

  // ---- one relative-attention block ----
  auto run_attn = [&](int pi, const bf16_t* qA, const bf16_t* kA, const bf16_t* vA,
                      const float* baseF, float* outF, bf16_t* outB) {
    const float *wq = IN[pi + 0], *bq = IN[pi + 1];
    const float *wk = IN[pi + 2], *bk = IN[pi + 3];
    const float *wv = IN[pi + 4], *bv = IN[pi + 5];
    const float *wo = IN[pi + 6], *bo = IN[pi + 7];
    const float *wr = IN[pi + 8], *br = IN[pi + 9];
    const float *u  = IN[pi + 10], *vp = IN[pi + 11];
    const float *g  = IN[pi + 12], *be = IN[pi + 13];

    // q projection (f32 out; +u / +v broadcast, cast to bf16)
    wt_kernel<<<ew((size_t)DM * DM), 256, 0, stream>>>(wq, wT);
    gemm(qA, 0, 0, DM, wT, 0, 0, DM, bq, qf, nullptr, 0, 0, DM,
         (int)TOK, DM, DM, 1, 1, 0);
    row_bias_cast_kernel<<<ew(TOK * DM), 256, 0, stream>>>(qf, u, qu, TOK * DM);
    row_bias_cast_kernel<<<ew(TOK * DM), 256, 0, stream>>>(qf, vp, qvb, TOK * DM);
    // k projection (bf16 out)
    wt_kernel<<<ew((size_t)DM * DM), 256, 0, stream>>>(wk, wT);
    gemm(kA, 0, 0, DM, wT, 0, 0, DM, bk, nullptr, kb, 0, 0, DM,
         (int)TOK, DM, DM, 1, 1, 0);
    // v projection (bf16), then transpose to [b,h,d,k]
    wt_kernel<<<ew((size_t)DM * DM), 256, 0, stream>>>(wv, wT);
    gemm(vA, 0, 0, DM, wT, 0, 0, DM, bv, nullptr, vb, 0, 0, DM,
         (int)TOK, DM, DM, 1, 1, 0);
    vtr_kernel<<<ew(TOK * DM), 256, 0, stream>>>(vb, vt);
    // r projection: 65536 x 512 x 512 (dominant GEMM)
    wt_kernel<<<ew((size_t)DM * DM), 256, 0, stream>>>(wr, wT);
    gemm(posw, 0, 0, DM, wT, 0, 0, DM, br, nullptr, rbuf, 0, 0, DM,
         SL * SL, DM, DM, 1, 1, 0);
    // AC scores: per (b,h) 256x256x64
    gemm(qu, (long long)SL * DM, DHD, DM,
         kb, (long long)SL * DM, DHD, DM,
         nullptr, sc, nullptr,
         (long long)NH * SL * SL, (long long)SL * SL, SL,
         SL, SL, DHD, NB, NH, 0);
    // BD scores (added)
    bd_kernel<<<dim3(SL, NH, SL / 64), 32, 0, stream>>>(qvb, rbuf, sc);
    // softmax -> bf16 attn
    softmax_kernel<<<dim3((unsigned)(NB * NH * SL / 8)), 256, 0, stream>>>(sc, at);
    // attn @ v : per (b,h) 256x64x256 -> ctx bf16 [b,q,h,d]
    gemm(at, (long long)NH * SL * SL, (long long)SL * SL, SL,
         vt, (long long)NH * DHD * SL, (long long)DHD * SL, SL,
         nullptr, nullptr, ctx,
         (long long)SL * DM, DHD, DM,
         SL, DHD, SL, NB, NH, 0);
    // output projection + residual + layernorm
    wt_kernel<<<ew((size_t)DM * DM), 256, 0, stream>>>(wo, wT);
    gemm(ctx, 0, 0, DM, wT, 0, 0, DM, bo, of, nullptr, 0, 0, DM,
         (int)TOK, DM, DM, 1, 1, 0);
    ln_kernel<<<dim3((unsigned)(TOK / 8)), 256, 0, stream>>>(baseF, of, g, be,
                                                             outF, outB);
  };

  // attn1: self (q=k=v=dec_input, base=dec_input)
  run_attn(4, decb, decb, decb, IN[0], x1f, x1b);
  // attn2: pos (q=k=dp, v=x1, base=x1 since values_provided)
  run_attn(18, dpb, dpb, x1b, x1f, x2f, x2b);
  // attn3: cross (q=x2, k=v=enc, base=x2)
  run_attn(32, x2b, encb, encb, x2f, x3f, nullptr);

  // ---- conv FFN: two K=9 convs as 9 accumulated per-tap GEMMs ----
  convpack_kernel<<<ew((size_t)KW * DM * DM), 256, 0, stream>>>(IN[46], w1p);
  convpack_kernel<<<ew((size_t)KW * DM * DM), 256, 0, stream>>>(IN[48], w2p);
  padcast_kernel<<<ew((size_t)NB * (SL + 2 * PD) * DM), 256, 0, stream>>>(x3f, xpad, 0);
  for (int t = 0; t < KW; ++t)
    gemm(xpad + (size_t)t * DM, (long long)(SL + 2 * PD) * DM, 0, DM,
         w1p + (size_t)t * DM * DM, 0, 0, DM,
         (t == 0) ? IN[47] : nullptr, h1f, nullptr,
         (long long)SL * DM, 0, DM,
         SL, DM, DM, NB, 1, (t == 0) ? 0 : 1);
  padcast_kernel<<<ew((size_t)NB * (SL + 2 * PD) * DM), 256, 0, stream>>>(h1f, h1p, 1);
  for (int t = 0; t < KW; ++t)
    gemm(h1p + (size_t)t * DM, (long long)(SL + 2 * PD) * DM, 0, DM,
         w2p + (size_t)t * DM * DM, 0, 0, DM,
         (t == 0) ? IN[49] : nullptr, h2f, nullptr,
         (long long)SL * DM, 0, DM,
         SL, DM, DM, NB, 1, (t == 0) ? 0 : 1);
  // final residual + layernorm -> d_out (fp32)
  ln_kernel<<<dim3((unsigned)(TOK / 8)), 256, 0, stream>>>(x3f, h2f, IN[50], IN[51],
                                                           (float*)d_out, nullptr);
}
